// HGCA_10857677324785
// MI455X (gfx1250) — matrix operand order repaired and checked by reading, not compiled, not run on hardware
//
#include <hip/hip_runtime.h>

// ---------- types for WMMA ----------
typedef __attribute__((ext_vector_type(16))) __bf16 bf16x16;
typedef __attribute__((ext_vector_type(8)))  __bf16 bf16x8;
typedef __attribute__((ext_vector_type(8)))  float  f32x8;

#define TAU_INV 2.0f   // 1/0.5
#define DIM 128
#define CHUNK 64       // columns staged per LDS buffer
#define BUFBYTES (2 * CHUNK * DIM * 2)   // hi+lo panel bytes per buffer (32KB)

// ---------------------------------------------------------------------
// CDNA5 async global->LDS copy (ASYNCcnt-tracked), inline asm so it is
// portable across both toolchains' builtin arities.
// ---------------------------------------------------------------------
__device__ __forceinline__ void async_cp16(unsigned lds_byte, const void* gaddr)
{
    asm volatile("global_load_async_to_lds_b128 %0, %1, off"
                 :: "v"(lds_byte), "v"(gaddr) : "memory");
}
__device__ __forceinline__ void wait_async0()
{
    asm volatile("s_wait_asynccnt 0" ::: "memory");
}

// =====================================================================
// Kernel 1: h = elu(z@W1+b1)@W2+b2 ; rowwise L2 normalize ; split into
// bf16 hi/lo pair.  64 rows per 256-thread block (1% of FLOPs, exact
// fp32 path).
// =====================================================================
__global__ __launch_bounds__(256)
void proj_elu_norm(const float* __restrict__ z,
                   const float* __restrict__ W1, const float* __restrict__ b1,
                   const float* __restrict__ W2, const float* __restrict__ b2,
                   __bf16* __restrict__ out_hi, __bf16* __restrict__ out_lo)
{
    __shared__ float sA[64 * DIM];
    __shared__ float sB[64 * DIM];
    __shared__ float sP[64][4];
    __shared__ float sInv[64];

    const int tid = threadIdx.x;
    const size_t i0 = (size_t)blockIdx.x * 64;

    #pragma unroll
    for (int e = 0; e < 32; ++e) {
        int flat = e * 256 + tid;
        sA[flat] = z[i0 * DIM + flat];
    }
    __syncthreads();

    #pragma unroll 1
    for (int e = 0; e < 32; ++e) {
        int flat = e * 256 + tid;
        int r = flat >> 7, c = flat & 127;
        float acc = b1[c];
        #pragma unroll 8
        for (int k = 0; k < DIM; ++k)
            acc = fmaf(sA[r * DIM + k], W1[k * DIM + c], acc);
        sB[flat] = acc > 0.0f ? acc : (expf(acc) - 1.0f);
    }
    __syncthreads();

    #pragma unroll 1
    for (int e = 0; e < 32; ++e) {
        int flat = e * 256 + tid;
        int r = flat >> 7, c = flat & 127;
        float acc = b2[c];
        #pragma unroll 8
        for (int k = 0; k < DIM; ++k)
            acc = fmaf(sB[r * DIM + k], W2[k * DIM + c], acc);
        sA[flat] = acc;
    }
    __syncthreads();

    {
        int r = tid >> 2, q = tid & 3;
        float ss = 0.0f;
        #pragma unroll
        for (int c = 0; c < 32; ++c) {
            float x = sA[r * DIM + q * 32 + c];
            ss = fmaf(x, x, ss);
        }
        sP[r][q] = ss;
    }
    __syncthreads();
    if (tid < 64) {
        float n = sqrtf(sP[tid][0] + sP[tid][1] + sP[tid][2] + sP[tid][3]);
        sInv[tid] = 1.0f / fmaxf(n, 1e-12f);
    }
    __syncthreads();

    #pragma unroll
    for (int e = 0; e < 32; ++e) {
        int flat = e * 256 + tid;
        int r = flat >> 7;
        float x  = sA[flat] * sInv[r];
        __bf16 h = (__bf16)x;
        float lo = x - (float)h;
        out_hi[i0 * DIM + flat] = h;
        out_lo[i0 * DIM + flat] = (__bf16)lo;
    }
}

// =====================================================================
// Kernel 2: rowsum(exp(A@B^T / tau)) for a 128-row strip of A vs all
// columns of B.  Split-bf16 (hi*hi + hi*lo + lo*hi), fp32 accum, dual
// accumulators per tile.  B panel staged to LDS via async-to-LDS DMA,
// double buffered, fragment-linear layout (ds_load_b128 at base+lane*32).
// One barrier per chunk.
// =====================================================================
__device__ __forceinline__ f32x8 wmma_bf16(bf16x16 a, bf16x16 b, f32x8 c)
{
    return __builtin_amdgcn_wmma_f32_16x16x32_bf16(
        false, a, false, b, (short)0, c, false, false);
}

__global__ __launch_bounds__(256)
void sim_rowsum(const __bf16* __restrict__ Ahi, const __bf16* __restrict__ Alo,
                const __bf16* __restrict__ Bhi, const __bf16* __restrict__ Blo,
                float* __restrict__ rowsum, int N)
{
    __shared__ __align__(32) unsigned char sMem[2][BUFBYTES];   // 2 x 32KB

    const int tid  = threadIdx.x;
    const int wave = tid >> 5;          // 0..7
    const int lane = tid & 31;
    const int half = lane >> 4;
    const int m16  = lane & 15;
    const int i0   = blockIdx.x * 128;
    const int rowBase = i0 + wave * 16;

    // ---- A fragments (16-bit A layout): elems 0..7 = K k0+half*8..,
    //      elems 8..15 = K k0+16+half*8 ; lane m16 = row ----
    bf16x16 afrag[2][4];
    {
        const __bf16* rh = Ahi + (size_t)(rowBase + m16) * DIM;
        const __bf16* rl = Alo + (size_t)(rowBase + m16) * DIM;
        #pragma unroll
        for (int ks = 0; ks < 4; ++ks) {
            int k0 = ks * 32;
            bf16x8 h0 = *(const bf16x8*)(rh + k0 + half * 8);
            bf16x8 h1 = *(const bf16x8*)(rh + k0 + 16 + half * 8);
            bf16x8 l0 = *(const bf16x8*)(rl + k0 + half * 8);
            bf16x8 l1 = *(const bf16x8*)(rl + k0 + 16 + half * 8);
            bf16x16 fh, fl;
            #pragma unroll
            for (int e = 0; e < 8; ++e) {
                fh[e] = h0[e]; fh[e + 8] = h1[e];
                fl[e] = l0[e]; fl[e + 8] = l1[e];
            }
            afrag[0][ks] = fh;
            afrag[1][ks] = fl;
        }
    }

    const unsigned char* sGen = &sMem[0][0];
    const unsigned ldsBase = (unsigned)(size_t)sGen;

    // chunk-invariant staging offsets: 1024 16B units per matrix,
    // 4 per thread per matrix.
    unsigned sLoff[4];
    unsigned sGoff[4];   // element offset within panel row block
    #pragma unroll
    for (int e = 0; e < 4; ++e) {
        int id = e * 256 + tid;               // 0..1023
        int r  = id >> 4, u = id & 15;        // row in chunk, 16B unit
        int nt = r >> 4, mm = r & 15;
        int ks = u >> 2, hh = u & 1, w = (u >> 1) & 1;
        sLoff[e] = (unsigned)(((nt * 4 + ks) << 10) + ((hh << 4) + mm) * 32 + w * 16);
        sGoff[e] = (unsigned)(r * DIM + u * 8);
    }

    auto stage = [&](int j0, int buf) {
        const unsigned b = ldsBase + (unsigned)buf * BUFBYTES;
        const __bf16* gh = Bhi + (size_t)j0 * DIM;
        const __bf16* gl = Blo + (size_t)j0 * DIM;
        #pragma unroll
        for (int e = 0; e < 4; ++e) {
            async_cp16(b + sLoff[e],                  gh + sGoff[e]);
            async_cp16(b + (BUFBYTES / 2) + sLoff[e], gl + sGoff[e]);
        }
    };

    float racc[8] = {0,0,0,0,0,0,0,0};

    stage(0, 0);
    int cur = 0;
    for (int j0 = 0; j0 < N; j0 += CHUNK) {
        wait_async0();           // this wave's DMA into 'cur' complete
        __syncthreads();         // everyone's DMA done; prev reads of cur^1 done
        if (j0 + CHUNK < N) stage(j0 + CHUNK, cur ^ 1);

        const unsigned char* bh = sGen + cur * BUFBYTES;
        const unsigned char* bl = bh + (BUFBYTES / 2);
        #pragma unroll
        for (int nt = 0; nt < 4; ++nt) {
            f32x8 c0 = {0.f,0.f,0.f,0.f,0.f,0.f,0.f,0.f};
            f32x8 c1 = {0.f,0.f,0.f,0.f,0.f,0.f,0.f,0.f};
            #pragma unroll
            for (int ks = 0; ks < 4; ++ks) {
                unsigned foff = (unsigned)(((nt * 4 + ks) << 10) + lane * 32);
                bf16x16 bfh = *(const bf16x16*)(bh + foff);
                bf16x16 bfl = *(const bf16x16*)(bl + foff);
                if (ks & 2) {
                    c1 = wmma_bf16(afrag[0][ks], bfh, c1);   // hi*hi
                    c1 = wmma_bf16(afrag[0][ks], bfl, c1);   // hi*lo
                    c1 = wmma_bf16(afrag[1][ks], bfh, c1);   // lo*hi
                } else {
                    c0 = wmma_bf16(afrag[0][ks], bfh, c0);
                    c0 = wmma_bf16(afrag[0][ks], bfl, c0);
                    c0 = wmma_bf16(afrag[1][ks], bfh, c0);
                }
            }
            #pragma unroll
            for (int v = 0; v < 8; ++v)
                racc[v] += __expf((c0[v] + c1[v]) * TAU_INV);
        }
        cur ^= 1;
    }

    // butterfly reduce within each 16-lane half (C layout: lanes0-15 ->
    // rows rowBase+v, lanes16-31 -> rows rowBase+8+v)
    #pragma unroll
    for (int off = 8; off >= 1; off >>= 1) {
        #pragma unroll
        for (int v = 0; v < 8; ++v)
            racc[v] += __shfl_xor(racc[v], off, 32);
    }
    if (m16 == 0) {
        int base = rowBase + half * 8;
        #pragma unroll
        for (int v = 0; v < 8; ++v)
            rowsum[base + v] = racc[v];
    }
}

// =====================================================================
// Kernel 3: diag_i = h1n_i . h2n_i  (split-bf16 recombined in fp32).
// 4 threads per row.
// =====================================================================
__global__ __launch_bounds__(256)
void diag_dot(const __bf16* __restrict__ Ahi, const __bf16* __restrict__ Alo,
              const __bf16* __restrict__ Bhi, const __bf16* __restrict__ Blo,
              float* __restrict__ dg)
{
    const int idx = blockIdx.x * 256 + threadIdx.x;
    const int row = idx >> 2, q = idx & 3;
    const size_t base = (size_t)row * DIM + q * 32;
    float s = 0.0f;
    #pragma unroll
    for (int e = 0; e < 32; ++e) {
        float a = (float)Ahi[base + e] + (float)Alo[base + e];
        float b = (float)Bhi[base + e] + (float)Blo[base + e];
        s = fmaf(a, b, s);
    }
    s += __shfl_xor(s, 1, 32);
    s += __shfl_xor(s, 2, 32);
    if (q == 0) dg[row] = s;
}

// =====================================================================
// Kernel 4: loss_i = 0.5*(log den1 + log den2) - diag_i / tau ; mean.
// =====================================================================
__global__ __launch_bounds__(256)
void finalize_loss(const float* __restrict__ rs11, const float* __restrict__ rs22,
                   const float* __restrict__ rs12, const float* __restrict__ rs21,
                   const float* __restrict__ dg, float* __restrict__ out, int N)
{
    __shared__ float red[256];
    const int tid = threadIdx.x;
    const float esq = expf(TAU_INV);   // exp(1/tau): refl diagonal
    float acc = 0.0f;
    for (int i = tid; i < N; i += 256) {
        float den1 = rs11[i] + rs12[i] - esq;
        float den2 = rs22[i] + rs21[i] - esq;
        acc += 0.5f * (logf(den1) + logf(den2)) - dg[i] * TAU_INV;
    }
    red[tid] = acc;
    __syncthreads();
    for (int s = 128; s > 0; s >>= 1) {
        if (tid < s) red[tid] += red[tid + s];
        __syncthreads();
    }
    if (tid == 0) out[0] = red[0] / (float)N;
}

// =====================================================================
extern "C" void kernel_launch(void* const* d_in, const int* in_sizes, int n_in,
                              void* d_out, int out_size, void* d_ws, size_t ws_size,
                              hipStream_t stream)
{
    const float* z1 = (const float*)d_in[0];
    const float* z2 = (const float*)d_in[1];
    const float* W1 = (const float*)d_in[2];
    const float* b1 = (const float*)d_in[3];
    const float* W2 = (const float*)d_in[4];
    const float* b2 = (const float*)d_in[5];
    const int N = in_sizes[0] / DIM;    // 16384

    char* ws = (char*)d_ws;
    const size_t mat = (size_t)N * DIM * sizeof(unsigned short);
    __bf16* h1hi = (__bf16*)(ws);
    __bf16* h1lo = (__bf16*)(ws + mat);
    __bf16* h2hi = (__bf16*)(ws + 2 * mat);
    __bf16* h2lo = (__bf16*)(ws + 3 * mat);
    float* rs11 = (float*)(ws + 4 * mat);
    float* rs22 = rs11 + N;
    float* rs12 = rs22 + N;
    float* rs21 = rs12 + N;
    float* dg   = rs21 + N;

    proj_elu_norm<<<N / 64, 256, 0, stream>>>(z1, W1, b1, W2, b2, h1hi, h1lo);
    proj_elu_norm<<<N / 64, 256, 0, stream>>>(z2, W1, b1, W2, b2, h2hi, h2lo);

    sim_rowsum<<<N / 128, 256, 0, stream>>>(h1hi, h1lo, h1hi, h1lo, rs11, N);
    sim_rowsum<<<N / 128, 256, 0, stream>>>(h2hi, h2lo, h2hi, h2lo, rs22, N);
    sim_rowsum<<<N / 128, 256, 0, stream>>>(h1hi, h1lo, h2hi, h2lo, rs12, N);
    sim_rowsum<<<N / 128, 256, 0, stream>>>(h2hi, h2lo, h1hi, h1lo, rs21, N);

    diag_dot<<<N * 4 / 256, 256, 0, stream>>>(h1hi, h1lo, h2hi, h2lo, dg);

    finalize_loss<<<1, 256, 0, stream>>>(rs11, rs22, rs12, rs21, dg, (float*)d_out, N);
}
